// MoE_65214783422490
// MI455X (gfx1250) — compile-verified, matrix operand order checked by reading, not run on hardware
//
#include <hip/hip_runtime.h>
#include <hip/hip_bf16.h>
#include <math.h>
#include <stdint.h>

#define BTOK 32768
#define DDIM 512
#define HDIM 2048
#define ODIM 512
#define NEXP 8
#define LN_EPS 1e-5f

typedef unsigned short u16;
typedef __bf16 bf16;
typedef bf16 v16bf __attribute__((ext_vector_type(16)));
typedef u16  v16u  __attribute__((ext_vector_type(16)));
typedef float v8f  __attribute__((ext_vector_type(8)));
typedef int  v4i   __attribute__((ext_vector_type(4)));

// ---- CDNA5 async global->LDS copy availability probe ----
#define HAVE_ASYNC 0
#if defined(__has_builtin)
#  if __has_builtin(__builtin_amdgcn_global_load_async_to_lds_b128)
#    undef HAVE_ASYNC
#    define HAVE_ASYNC 1
#  endif
#endif
#define AS1 __attribute__((address_space(1)))
#define AS3 __attribute__((address_space(3)))

__device__ __forceinline__ void wait_async_all() {
#if HAVE_ASYNC
#  if __has_builtin(__builtin_amdgcn_s_wait_asynccnt)
    __builtin_amdgcn_s_wait_asynccnt(0);
#  else
    asm volatile("s_wait_asynccnt 0" ::: "memory");
#  endif
#endif
}

// async copy of one 16-byte chunk: global (AS1) -> LDS (AS3), ASYNCcnt-tracked
__device__ __forceinline__ void async_cp16(const u16* gsrc, u16* ldst) {
#if HAVE_ASYNC
    __builtin_amdgcn_global_load_async_to_lds_b128(
        (AS1 v4i*)(uintptr_t)gsrc,
        (AS3 v4i*)(unsigned)(uintptr_t)ldst, 0, 0);
#else
    *(uint4*)ldst = *(const uint4*)gsrc;
#endif
}

// ---------- bf16 helpers (RNE, bit ops only) ----------
__device__ __forceinline__ u16 f2bf(float f) {
    unsigned u = __float_as_uint(f);
    return (u16)((u + 0x7FFFu + ((u >> 16) & 1u)) >> 16);
}
__device__ __forceinline__ float bf2f(u16 h) {
    return __uint_as_float(((unsigned)h) << 16);
}

// ---------- WMMA fragment loader ----------
// A (row-major over K) and B (weights pre-transposed to [N][K]) share this
// layout.  ISA 7.12.2 (16-bit, 16x32): element j of lane l holds
// K = (j>>3)*16 + (l>>4)*8 + (j&7), row/col = l&15.  Each 8-element group is
// contiguous -> lowers to b128 loads.
__device__ __forceinline__ v16bf load_frag(const u16* __restrict__ base, int ld,
                                           int nrow, int k0, int lane) {
    const u16* p = base + (size_t)(nrow + (lane & 15)) * ld + k0 + ((lane >> 4) << 3);
    v16u v;
#pragma unroll
    for (int g = 0; g < 2; ++g)
#pragma unroll
        for (int q = 0; q < 8; ++q)
            v[g * 8 + q] = p[g * 16 + q];
    return __builtin_bit_cast(v16bf, v);
}

// ---------- init: zero out / counters ----------
__global__ void moe_init(float* __restrict__ out, int* __restrict__ cnt,
                         float* __restrict__ probsum) {
    size_t i = (size_t)blockIdx.x * blockDim.x + threadIdx.x;
    size_t stride = (size_t)gridDim.x * blockDim.x;
    size_t n = (size_t)BTOK * ODIM;
    for (size_t j = i; j < n; j += stride) out[j] = 0.f;
    if (i < NEXP) { cnt[i] = 0; probsum[i] = 0.f; }
}

// ---------- weight convert + transpose to bf16 ----------
__global__ void conv_w1(const float* __restrict__ W1, u16* __restrict__ w1t) {
    size_t n = (size_t)NEXP * DDIM * HDIM;
    size_t stride = (size_t)gridDim.x * blockDim.x;
    for (size_t i = (size_t)blockIdx.x * blockDim.x + threadIdx.x; i < n; i += stride) {
        size_t e = i / ((size_t)DDIM * HDIM);
        size_t rem = i % ((size_t)DDIM * HDIM);
        size_t h = rem / DDIM, d = rem % DDIM;
        w1t[i] = f2bf(W1[e * DDIM * HDIM + d * HDIM + h]);   // w1t[e][h][d]
    }
}
__global__ void conv_w2(const float* __restrict__ W2, u16* __restrict__ w2t) {
    size_t n = (size_t)NEXP * HDIM * ODIM;
    size_t stride = (size_t)gridDim.x * blockDim.x;
    for (size_t i = (size_t)blockIdx.x * blockDim.x + threadIdx.x; i < n; i += stride) {
        size_t e = i / ((size_t)HDIM * ODIM);
        size_t rem = i % ((size_t)HDIM * ODIM);
        size_t o = rem / HDIM, h = rem % HDIM;
        w2t[i] = f2bf(W2[e * HDIM * ODIM + h * ODIM + o]);   // w2t[e][o][h]
    }
}

// ---------- gating + LN + routing: one wave32 per token ----------
__global__ __launch_bounds__(256)
void moe_gate(const float* __restrict__ x, const float* __restrict__ gw,
              const float* __restrict__ gb, u16* __restrict__ xnbf,
              float* __restrict__ comb, int* __restrict__ lists,
              int* __restrict__ cnt, float* __restrict__ probsum) {
    const int lane = threadIdx.x & 31;
    const int wv   = threadIdx.x >> 5;
    const int t = blockIdx.x * 8 + wv;
    if (t >= BTOK) return;
    const float* xr = x + (size_t)t * DDIM;

    float xv[16];
    float s = 0.f, ss = 0.f;
    float lg[NEXP];
#pragma unroll
    for (int e = 0; e < NEXP; ++e) lg[e] = 0.f;

#pragma unroll
    for (int i = 0; i < 16; ++i) {
        int d = lane + 32 * i;
        float v = xr[d];
        xv[i] = v;
        s += v; ss += v * v;
        const float4* g4 = (const float4*)(gw + (size_t)d * NEXP);
        float4 a = g4[0], b = g4[1];
        lg[0] += v * a.x; lg[1] += v * a.y; lg[2] += v * a.z; lg[3] += v * a.w;
        lg[4] += v * b.x; lg[5] += v * b.y; lg[6] += v * b.z; lg[7] += v * b.w;
    }
#pragma unroll
    for (int off = 16; off >= 1; off >>= 1) {
        s  += __shfl_xor(s,  off, 32);
        ss += __shfl_xor(ss, off, 32);
#pragma unroll
        for (int e = 0; e < NEXP; ++e) lg[e] += __shfl_xor(lg[e], off, 32);
    }
    float mu   = s / (float)DDIM;
    float var  = ss / (float)DDIM - mu * mu;
    float rsig = rsqrtf(var + LN_EPS);
#pragma unroll
    for (int i = 0; i < 16; ++i) {
        int d = lane + 32 * i;
        xnbf[(size_t)t * DDIM + d] = f2bf((xv[i] - mu) * rsig);
    }
    // softmax over experts
#pragma unroll
    for (int e = 0; e < NEXP; ++e) lg[e] += gb[e];
    float mx = lg[0];
#pragma unroll
    for (int e = 1; e < NEXP; ++e) mx = fmaxf(mx, lg[e]);
    float den = 0.f;
#pragma unroll
    for (int e = 0; e < NEXP; ++e) { lg[e] = __expf(lg[e] - mx); den += lg[e]; }
    float inv = 1.f / den;
#pragma unroll
    for (int e = 0; e < NEXP; ++e) lg[e] *= inv;
    // top-2 (strict >, lowest index wins ties, matching lax.top_k)
    int i1 = 0; float m1 = lg[0];
#pragma unroll
    for (int e = 1; e < NEXP; ++e) if (lg[e] > m1) { m1 = lg[e]; i1 = e; }
    int i2 = -1; float m2 = -1.f;
#pragma unroll
    for (int e = 0; e < NEXP; ++e) if (e != i1 && lg[e] > m2) { m2 = lg[e]; i2 = e; }

    if (lane == 0) {
#pragma unroll
        for (int e = 0; e < NEXP; ++e)
            comb[(size_t)t * NEXP + e] = (e == i1) ? m1 : ((e == i2) ? m2 : 0.f);
#pragma unroll
        for (int e = 0; e < NEXP; ++e) atomicAdd(&probsum[e], lg[e]);
        int p1 = atomicAdd(&cnt[i1], 1); lists[(size_t)i1 * BTOK + p1] = t;
        int p2 = atomicAdd(&cnt[i2], 1); lists[(size_t)i2 * BTOK + p2] = t;
    }
}

// ---------- load balance loss ----------
__global__ void moe_loss(const float* __restrict__ probsum, float* __restrict__ out_loss) {
    float acc = 0.f;
#pragma unroll
    for (int e = 0; e < NEXP; ++e) {
        float p = probsum[e] / (float)BTOK;
        acc += p * p;
    }
    *out_loss = acc * (float)NEXP;
}

// ---------- W1 panel staging: 64 h-rows x 512 d, bf16 ----------
// Async path: per wave, rows [16w,16w+16); each row = 1024 B = 2 async b128
// transfers of 512 B (32 lanes x 16 B).  Tracked on ASYNCcnt, no VGPR staging.
__device__ __forceinline__ void stage_w1_panel(const u16* __restrict__ gsrc,
                                               u16* __restrict__ lds,
                                               int lda, int wid, int lane) {
#if HAVE_ASYNC
#pragma unroll 4
    for (int r = 0; r < 16; ++r) {
        int row = wid * 16 + r;
        const u16* g = gsrc + (size_t)row * DDIM + lane * 8;  // 16 B / lane
        u16* l = lds + row * lda + lane * 8;
        async_cp16(g, l);
        async_cp16(g + 256, l + 256);
    }
#else
    // synchronous fallback: 16 B chunks, 64 chunks per row
    for (int i = threadIdx.x; i < 64 * 64; i += 128) {
        int row = i >> 6, c = i & 63;
        ((uint4*)(lds + row * lda))[c] = ((const uint4*)(gsrc + (size_t)row * DDIM))[c];
    }
#endif
}

// ---------- routed expert FFN: fused GEMM1 -> GELU -> GEMM2 ----------
// Block = 128 threads = 4 waves. WG tile: 64 tokens x 64 outputs.
#define LDA 520   // 512 + 8 pad (u16 units)
#define LDH 72    // 64 + 8 pad
__global__ __launch_bounds__(128)
void moe_ffn(const u16* __restrict__ xnbf, const u16* __restrict__ w1t,
             const u16* __restrict__ w2t, const float* __restrict__ lnw,
             const float* __restrict__ lnb, const float* __restrict__ b1,
             const float* __restrict__ b2, const float* __restrict__ comb,
             const int* __restrict__ lists, const int* __restrict__ cnt,
             float* __restrict__ out) {
    const int e    = blockIdx.z;
    const int tile = blockIdx.y;
    const int n0   = blockIdx.x * 64;
    const int ce   = cnt[e];
    if (tile * 64 >= ce) return;

    extern __shared__ char smem_raw[];
    u16*   aLds   = (u16*)smem_raw;                               // 64 x LDA
    u16*   w1p    = (u16*)(smem_raw + 64 * LDA * 2);              // 64 x LDA
    u16*   hLds   = (u16*)(smem_raw + 2 * 64 * LDA * 2);          // 64 x LDH
    int*   tokLds = (int*)(smem_raw + 2 * 64 * LDA * 2 + 64 * LDH * 2);
    float* wLds   = (float*)(tokLds + 64);

    const int tid = threadIdx.x;
    const int lane = tid & 31, wid = tid >> 5;

    if (tid < 64) {
        int idx = tile * 64 + tid;
        int t = (idx < ce) ? lists[(size_t)e * BTOK + idx] : -1;
        tokLds[tid] = t;
        wLds[tid] = (t >= 0) ? comb[(size_t)t * NEXP + e] : 0.f;
    }
    __syncthreads();

    // stage A = (xn * ln_w[e] + ln_b[e]) as bf16
    const float* lw = lnw + (size_t)e * DDIM;
    const float* lb = lnb + (size_t)e * DDIM;
    for (int i = tid; i < 64 * DDIM; i += 128) {
        int r = i >> 9, d = i & (DDIM - 1);
        int t = tokLds[r];
        float v = 0.f;
        if (t >= 0) v = bf2f(xnbf[(size_t)t * DDIM + d]) * lw[d] + lb[d];
        aLds[r * LDA + d] = f2bf(v);
    }

    const u16* w1e = w1t + (size_t)e * HDIM * DDIM;  // [H][D]
    const u16* w2e = w2t + (size_t)e * ODIM * HDIM;  // [O][H]
    const float* b1e = b1 + (size_t)e * HDIM;
    const float* b2e = b2 + (size_t)e * ODIM;

    // kick off first W1 panel copy (overlaps with A staging above)
    stage_w1_panel(w1e, w1p, LDA, wid, lane);

    const v8f vz = {};
    v8f outC[4];
#pragma unroll
    for (int i = 0; i < 4; ++i) outC[i] = vz;

    for (int hc = 0; hc < HDIM; hc += 64) {
        wait_async_all();
        __syncthreads();                 // panel(hc) + aLds visible to all waves

        // ---- GEMM1: hC = A[64xD] * W1panel[64][D]^T (both LDS) ----
        v8f hC[4];
#pragma unroll
        for (int i = 0; i < 4; ++i) hC[i] = vz;
        for (int kk = 0; kk < DDIM; kk += 32) {
            v16bf a = load_frag(aLds, LDA, wid * 16, kk, lane);
#pragma unroll
            for (int nt = 0; nt < 4; ++nt) {
                v16bf b = load_frag(w1p, LDA, nt * 16, kk, lane);
                hC[nt] = __builtin_amdgcn_wmma_f32_16x16x32_bf16(
                    false, a, false, b, (short)0, hC[nt], false, false);
            }
        }
        __syncthreads();                 // all waves done reading panel(hc)

        // prefetch next panel asynchronously; overlaps with GELU + GEMM2
        if (hc + 64 < HDIM)
            stage_w1_panel(w1e + (size_t)(hc + 64) * DDIM, w1p, LDA, wid, lane);

        // ---- bias + exact GELU -> bf16 h chunk in LDS ----
#pragma unroll
        for (int nt = 0; nt < 4; ++nt)
#pragma unroll
            for (int r = 0; r < 8; ++r) {
                int row = wid * 16 + r + ((lane >> 4) << 3);
                int col = nt * 16 + (lane & 15);
                float pre = hC[nt][r] + b1e[hc + col];
                float g = 0.5f * pre * (1.f + erff(pre * 0.70710678118f));
                hLds[row * LDH + col] = f2bf(g);
            }
        __syncthreads();

        // ---- GEMM2: outC += h[64x64] * W2t[n0:n0+64][hc:hc+64]^T ----
        if (hc + 64 < HDIM)
            __builtin_prefetch(w2e + (size_t)n0 * HDIM + hc + 64, 0, 1);
#pragma unroll
        for (int k2 = 0; k2 < 64; k2 += 32) {
            v16bf a = load_frag(hLds, LDH, wid * 16, k2, lane);
#pragma unroll
            for (int nt = 0; nt < 4; ++nt) {
                v16bf b = load_frag(w2e, HDIM, n0 + nt * 16, hc + k2, lane);
                outC[nt] = __builtin_amdgcn_wmma_f32_16x16x32_bf16(
                    false, a, false, b, (short)0, outC[nt], false, false);
            }
        }
        __syncthreads();                 // hLds reads done before next overwrite
    }

    // ---- weighted scatter into out (K=2 experts merge via f32 atomics) ----
#pragma unroll
    for (int nt = 0; nt < 4; ++nt)
#pragma unroll
        for (int r = 0; r < 8; ++r) {
            int row = wid * 16 + r + ((lane >> 4) << 3);
            int col = n0 + nt * 16 + (lane & 15);
            float w = wLds[row];
            if (w != 0.f) {
                int t = tokLds[row];
                atomicAdd(out + (size_t)t * ODIM + col,
                          w * (outC[nt][r] + b2e[col]));
            }
        }
}

extern "C" void kernel_launch(void* const* d_in, const int* in_sizes, int n_in,
                              void* d_out, int out_size, void* d_ws, size_t ws_size,
                              hipStream_t stream) {
    const float* x    = (const float*)d_in[0];
    const float* gw   = (const float*)d_in[1];
    const float* gb   = (const float*)d_in[2];
    const float* lnw  = (const float*)d_in[3];
    const float* lnb  = (const float*)d_in[4];
    const float* W1   = (const float*)d_in[5];
    const float* b1   = (const float*)d_in[6];
    const float* W2   = (const float*)d_in[7];
    const float* b2   = (const float*)d_in[8];
    float* out = (float*)d_out;

    char* ws = (char*)d_ws;
    u16* xnbf = (u16*)ws;            ws += (size_t)BTOK * DDIM * 2;        // 32 MB
    u16* w1t  = (u16*)ws;            ws += (size_t)NEXP * HDIM * DDIM * 2; // 16 MB
    u16* w2t  = (u16*)ws;            ws += (size_t)NEXP * ODIM * HDIM * 2; // 16 MB
    float* comb = (float*)ws;        ws += (size_t)BTOK * NEXP * 4;        // 1 MB
    int* lists  = (int*)ws;          ws += (size_t)NEXP * BTOK * 4;        // 1 MB
    int* cnt    = (int*)ws;          ws += 256;
    float* probsum = (float*)ws;     ws += 256;

    moe_init<<<4096, 256, 0, stream>>>(out, cnt, probsum);
    conv_w1<<<4096, 256, 0, stream>>>(W1, w1t);
    conv_w2<<<4096, 256, 0, stream>>>(W2, w2t);
    moe_gate<<<BTOK / 8, 256, 0, stream>>>(x, gw, gb, xnbf, comb, lists, cnt, probsum);
    moe_loss<<<1, 1, 0, stream>>>(probsum, out + (size_t)BTOK * ODIM);

    dim3 grid(ODIM / 64, (BTOK + 63) / 64, NEXP);  // (8, 512, 8); surplus tiles exit
    size_t smem = (size_t)2 * 64 * LDA * 2 + (size_t)64 * LDH * 2 + 64 * 4 + 64 * 4; // ~143 KB of 320 KB WGP LDS
    moe_ffn<<<grid, 128, smem, stream>>>(xnbf, w1t, w2t, lnw, lnb, b1, b2,
                                         comb, lists, cnt, out);
}
